// Llama2_27058293965583
// MI455X (gfx1250) — compile-verified, matrix operand order checked by reading, not run on hardware
//
#include <hip/hip_runtime.h>

// ---------------- constants (match reference) ----------------
constexpr int Lc  = 4;
constexpr int Dc  = 2048;
constexpr int Hc  = 16;
constexpr int HDc = 128;
constexpr int Fc  = 5632;
constexpr int Bc  = 4;
constexpr int Sc  = 1024;
constexpr int Mc  = Bc * Sc;          // 4096 token rows
constexpr float EPSc = 1e-5f;

typedef __bf16 bf16_t;
typedef __attribute__((ext_vector_type(16))) __bf16 v16bf;
typedef __attribute__((ext_vector_type(8)))  __bf16 v8bf;
typedef __attribute__((ext_vector_type(8)))  float  v8f;

// ---------------- WMMA helpers ----------------
// 16-bit A-matrix 16x32 fragment layout (wave32):
// lanes 0-15: row=lane, K = k0+0..7 (v0-3) and k0+16..23 (v4-7)
// lanes16-31: row=lane-16, K = k0+8..15 and k0+24..31
__device__ __forceinline__ v16bf frag_row(const bf16_t* base, int row, int ld,
                                          int k0, int lane) {
    const bf16_t* p = base + (size_t)(row + (lane & 15)) * ld + k0 + ((lane >> 4) << 3);
    union { v16bf v; v8bf h[2]; } u;
    u.h[0] = *(const v8bf*)(p);
    u.h[1] = *(const v8bf*)(p + 16);
    return u.v;
}

__device__ __forceinline__ v8f wmma_bf16(v16bf a, v16bf b, v8f c) {
    return __builtin_amdgcn_wmma_f32_16x16x32_bf16(
        /*neg_a=*/false, a, /*neg_b=*/false, b,
        /*c_mod=*/(short)0, c, /*reuse_a=*/false, /*reuse_b=*/false);
}

// ---------------- embedding gather ----------------
__global__ __launch_bounds__(256) void embed_gather(const int* __restrict__ ids,
                                                    const float* __restrict__ emb,
                                                    float* __restrict__ x) {
    int t = blockIdx.x * 256 + threadIdx.x;     // total Mc*Dc
    int c = t & (Dc - 1);
    int r = t >> 11;
    x[t] = emb[(size_t)ids[r] * Dc + c];
}

// ---------------- rmsnorm -> bf16 ----------------
__global__ __launch_bounds__(256) void rmsnorm_bf16(const float* __restrict__ x,
                                                    const float* __restrict__ w,
                                                    bf16_t* __restrict__ out) {
    __shared__ float red[256];
    int row = blockIdx.x;
    const float* xr = x + (size_t)row * Dc;
    float ss = 0.f;
    for (int c = threadIdx.x; c < Dc; c += 256) { float v = xr[c]; ss += v * v; }
    red[threadIdx.x] = ss;
    __syncthreads();
    for (int s = 128; s > 0; s >>= 1) {
        if (threadIdx.x < s) red[threadIdx.x] += red[threadIdx.x + s];
        __syncthreads();
    }
    float rs = rsqrtf(red[0] * (1.0f / Dc) + EPSc);
    for (int c = threadIdx.x; c < Dc; c += 256)
        out[(size_t)row * Dc + c] = (bf16_t)(xr[c] * rs * w[c]);
}

// ---------------- GEMM: C[MxN] (+)= A_bf16[MxK] * W_f32[KxN] ----------------
// block: 256 threads = 8 waves, tile 256x64, K step 32.
// Each wave owns 32 rows x 64 cols: 2 A-frags, 4 shared B-frags, 8 WMMAs/step.
// W tile staged to LDS as bf16, layout lds[col][k], pitch 40 (80B: 16B aligned,
// bank-conflict-free for b128 fragment reads).
constexpr int WPITCH = 40;

template <int ADD>
__global__ __launch_bounds__(256)
void gemm_bf16w32(const bf16_t* __restrict__ A, const float* __restrict__ W,
                  float* __restrict__ C, int M, int N, int K) {
    __shared__ bf16_t lw[64 * WPITCH];
    const int tid  = threadIdx.x;
    const int wave = tid >> 5;
    const int lane = tid & 31;
    const int row0 = blockIdx.y * 256 + wave * 32;
    const int col0 = blockIdx.x * 64;

    v8f acc[2][4];
    #pragma unroll
    for (int s = 0; s < 2; ++s)
        #pragma unroll
        for (int n = 0; n < 4; ++n) { v8f z = {}; acc[s][n] = z; }

    const int sk  = tid >> 3;         // 0..31 : k within tile
    const int scg = (tid & 7) * 8;    // 0..56 : col group of 8

    for (int k0 = 0; k0 < K; k0 += 32) {
        const float* wrow = W + (size_t)(k0 + sk) * N + col0 + scg;
        float4 w0 = *(const float4*)(wrow);
        float4 w1 = *(const float4*)(wrow + 4);
        __syncthreads();                       // prior frag reads done
        lw[(scg + 0) * WPITCH + sk] = (bf16_t)w0.x;
        lw[(scg + 1) * WPITCH + sk] = (bf16_t)w0.y;
        lw[(scg + 2) * WPITCH + sk] = (bf16_t)w0.z;
        lw[(scg + 3) * WPITCH + sk] = (bf16_t)w0.w;
        lw[(scg + 4) * WPITCH + sk] = (bf16_t)w1.x;
        lw[(scg + 5) * WPITCH + sk] = (bf16_t)w1.y;
        lw[(scg + 6) * WPITCH + sk] = (bf16_t)w1.z;
        lw[(scg + 7) * WPITCH + sk] = (bf16_t)w1.w;
        __syncthreads();

        v16bf a0 = frag_row(A, row0,      K, k0, lane);
        v16bf a1 = frag_row(A, row0 + 16, K, k0, lane);
        v16bf b[4];
        #pragma unroll
        for (int nt = 0; nt < 4; ++nt)
            b[nt] = frag_row(lw, nt * 16, WPITCH, 0, lane);
        #pragma unroll
        for (int nt = 0; nt < 4; ++nt) {
            acc[0][nt] = wmma_bf16(a0, b[nt], acc[0][nt]);
            acc[1][nt] = wmma_bf16(a1, b[nt], acc[1][nt]);
        }
    }

    const int ccol = lane & 15;
    const int rhi  = (lane >> 4) << 3;
    #pragma unroll
    for (int s = 0; s < 2; ++s) {
        #pragma unroll
        for (int nt = 0; nt < 4; ++nt) {
            #pragma unroll
            for (int r = 0; r < 8; ++r) {
                size_t idx = (size_t)(row0 + s * 16 + rhi + r) * N + col0 + nt * 16 + ccol;
                float v = acc[s][nt][r];
                if (ADD) C[idx] += v; else C[idx] = v;
            }
        }
    }
}

// ---------------- RoPE + pack Q/K to bf16 [B,H,S,HD] ----------------
__global__ __launch_bounds__(256) void rope_pack(const float* __restrict__ Qf,
                                                 const float* __restrict__ Kf,
                                                 bf16_t* __restrict__ Qb,
                                                 bf16_t* __restrict__ Kb) {
    int t = blockIdx.x * 256 + threadIdx.x;   // Bc*Sc*Hc*64 threads
    int d = t & 63;
    int h = (t >> 6) & (Hc - 1);
    int s = (t >> 10) & (Sc - 1);
    int b = t >> 20;
    float inv = __powf(10000.0f, -(float)d * (1.0f / 64.0f));
    float sn, cs;
    __sincosf((float)s * inv, &sn, &cs);
    size_t src = ((size_t)(b * Sc + s)) * Dc + h * HDc;
    size_t dst = ((size_t)((b * Hc + h) * Sc + s)) * HDc;
    float q1 = Qf[src + d], q2 = Qf[src + d + 64];
    float k1 = Kf[src + d], k2 = Kf[src + d + 64];
    Qb[dst + d]      = (bf16_t)(q1 * cs - q2 * sn);
    Qb[dst + d + 64] = (bf16_t)(q2 * cs + q1 * sn);
    Kb[dst + d]      = (bf16_t)(k1 * cs - k2 * sn);
    Kb[dst + d + 64] = (bf16_t)(k2 * cs + k1 * sn);
}

// ---------------- V transpose to bf16 [B,H,HD,S] ----------------
__global__ __launch_bounds__(256) void vpack(const float* __restrict__ Vf,
                                             bf16_t* __restrict__ Vt) {
    int t = blockIdx.x * 256 + threadIdx.x;   // Bc*Sc*Hc*HDc threads
    int d = t & (HDc - 1);
    int h = (t >> 7) & (Hc - 1);
    int s = (t >> 11) & (Sc - 1);
    int b = t >> 21;
    Vt[((size_t)((b * Hc + h) * HDc + d)) * Sc + s] =
        (bf16_t)Vf[((size_t)(b * Sc + s)) * Dc + h * HDc + d];
}

// ---------------- flash attention: 1 wave per 16-query tile ----------------
__global__ __launch_bounds__(256)
void attn_flash(const bf16_t* __restrict__ Qb, const bf16_t* __restrict__ Kb,
                const bf16_t* __restrict__ Vt, const int* __restrict__ amask,
                bf16_t* __restrict__ Ob) {
    __shared__ bf16_t pball[8 * 16 * WPITCH];
    const int wave = threadIdx.x >> 5;
    const int lane = threadIdx.x & 31;
    bf16_t* pb = pball + wave * 16 * WPITCH;   // wave-private P tile 16x32

    const int qt = blockIdx.x * 8 + wave;      // 0..63
    const int h  = blockIdx.y;
    const int b  = blockIdx.z;
    const int col = lane & 15;
    const int rhi = (lane >> 4) << 3;
    const int q0 = qt << 4;

    const bf16_t* Qh = Qb + ((size_t)(b * Hc + h)) * Sc * HDc;
    const bf16_t* Kh = Kb + ((size_t)(b * Hc + h)) * Sc * HDc;
    const bf16_t* Vh = Vt + ((size_t)(b * Hc + h)) * HDc * Sc;

    v16bf aq[4];
    #pragma unroll
    for (int kk = 0; kk < 4; ++kk) aq[kk] = frag_row(Qh, q0, HDc, kk * 32, lane);

    float rowmax[8], rowsum[8];
    v8f o[8];
    #pragma unroll
    for (int r = 0; r < 8; ++r) { rowmax[r] = -1e30f; rowsum[r] = 0.f; }
    #pragma unroll
    for (int n = 0; n < 8; ++n) { v8f z = {}; o[n] = z; }

    const int Lk = (((qt + 1) << 4) + 31) & ~31;   // keys, rounded to 32 (<=1024)
    const float scale = 0.0883883476483184f;       // 1/sqrt(128)

    for (int j0 = 0; j0 < Lk; j0 += 32) {
        v8f sa = {}, sb = {};
        #pragma unroll
        for (int kk = 0; kk < 4; ++kk) {
            sa = wmma_bf16(aq[kk], frag_row(Kh, j0,      HDc, kk * 32, lane), sa);
            sb = wmma_bf16(aq[kk], frag_row(Kh, j0 + 16, HDc, kk * 32, lane), sb);
        }
        const int ja = j0 + col, jb = j0 + 16 + col;
        const float ba  = (amask[(size_t)b * Sc + ja] != 0) ? 0.f : -1e9f;
        const float bbv = (amask[(size_t)b * Sc + jb] != 0) ? 0.f : -1e9f;
        #pragma unroll
        for (int r = 0; r < 8; ++r) {
            int i = q0 + rhi + r;
            float va = (ja <= i) ? sa[r] * scale + ba  : -1e9f;
            float vb = (jb <= i) ? sb[r] * scale + bbv : -1e9f;
            float t = fmaxf(va, vb);
            t = fmaxf(t, __shfl_xor(t, 1, 32));
            t = fmaxf(t, __shfl_xor(t, 2, 32));
            t = fmaxf(t, __shfl_xor(t, 4, 32));
            t = fmaxf(t, __shfl_xor(t, 8, 32));
            float nm   = fmaxf(rowmax[r], t);
            float corr = __expf(rowmax[r] - nm);
            rowmax[r] = nm;
            float pa = __expf(va - nm);
            float pv = __expf(vb - nm);
            rowsum[r] = rowsum[r] * corr + pa + pv;
            #pragma unroll
            for (int n = 0; n < 8; ++n) o[n][r] = o[n][r] * corr;
            pb[(rhi + r) * WPITCH + col]      = (bf16_t)pa;
            pb[(rhi + r) * WPITCH + col + 16] = (bf16_t)pv;
        }
        asm volatile("s_wait_dscnt 0" ::: "memory");   // P stores visible to frag read
        v16bf ap = frag_row(pb, 0, WPITCH, 0, lane);
        #pragma unroll
        for (int n = 0; n < 8; ++n)
            o[n] = wmma_bf16(ap, frag_row(Vh, n * 16, Sc, j0, lane), o[n]);
    }

    #pragma unroll
    for (int r = 0; r < 8; ++r) {
        float s = rowsum[r];
        s += __shfl_xor(s, 1, 32);
        s += __shfl_xor(s, 2, 32);
        s += __shfl_xor(s, 4, 32);
        s += __shfl_xor(s, 8, 32);
        float rcp = 1.0f / s;
        int srow = q0 + rhi + r;
        size_t base = ((size_t)(b * Sc + srow)) * Dc + h * HDc + col;
        #pragma unroll
        for (int n = 0; n < 8; ++n)
            Ob[base + n * 16] = (bf16_t)(o[n][r] * rcp);
    }
}

// ---------------- SiLU(g)*u -> bf16 ----------------
__global__ __launch_bounds__(256) void silu_mul(const float* __restrict__ G,
                                                const float* __restrict__ U,
                                                bf16_t* __restrict__ T) {
    int t = blockIdx.x * 256 + threadIdx.x;   // Mc*Fc threads
    float g = G[t];
    T[t] = (bf16_t)(g / (1.f + __expf(-g)) * U[t]);
}

// ---------------- final rmsnorm + head MLP (tiny) ----------------
__global__ __launch_bounds__(256)
void head_kernel(const float* __restrict__ x, const float* __restrict__ finw,
                 const float* __restrict__ catf,
                 const float* __restrict__ fc1w, const float* __restrict__ fc1b,
                 const float* __restrict__ fc2w, const float* __restrict__ fc2b,
                 const float* __restrict__ outw, const float* __restrict__ outb,
                 float* __restrict__ out) {
    __shared__ float feat[Dc + 10];
    __shared__ float red[256];
    __shared__ float z1[128];
    __shared__ float z2[64];
    const int t = threadIdx.x;
    for (int b = 0; b < Bc; ++b) {
        const float* xr = x + (size_t)b * Sc * Dc;   // s = 0 row
        float ss = 0.f;
        for (int c = t; c < Dc; c += 256) { float v = xr[c]; ss += v * v; }
        red[t] = ss;
        __syncthreads();
        for (int s = 128; s > 0; s >>= 1) {
            if (t < s) red[t] += red[t + s];
            __syncthreads();
        }
        float rs = rsqrtf(red[0] * (1.0f / Dc) + EPSc);
        for (int c = t; c < Dc; c += 256) feat[c] = xr[c] * rs * finw[c];
        if (t < 10) feat[Dc + t] = catf[b * 10 + t];
        __syncthreads();
        if (t < 128) {
            float a = fc1b[t];
            for (int i = 0; i < Dc + 10; ++i) a += feat[i] * fc1w[i * 128 + t];
            z1[t] = fmaxf(a, 0.f);
        }
        __syncthreads();
        if (t < 64) {
            float a = fc2b[t];
            for (int i = 0; i < 128; ++i) a += z1[i] * fc2w[i * 64 + t];
            z2[t] = fmaxf(a, 0.f);
        }
        __syncthreads();
        if (t == 0) {
            float a = outb[0];
            for (int i = 0; i < 64; ++i) a += z2[i] * outw[i];
            out[b] = a;
        }
        __syncthreads();
    }
}

// ---------------- host orchestration ----------------
extern "C" void kernel_launch(void* const* d_in, const int* in_sizes, int n_in,
                              void* d_out, int out_size, void* d_ws, size_t ws_size,
                              hipStream_t stream) {
    (void)in_sizes; (void)n_in; (void)out_size; (void)ws_size;
    const int*   ids   = (const int*)d_in[0];
    const int*   amask = (const int*)d_in[1];
    const float* catf  = (const float*)d_in[2];
    const float* emb   = (const float*)d_in[3];
    const float* Wq    = (const float*)d_in[4];
    const float* Wk    = (const float*)d_in[5];
    const float* Wv    = (const float*)d_in[6];
    const float* Wo    = (const float*)d_in[7];
    const float* Wg    = (const float*)d_in[8];
    const float* Wu    = (const float*)d_in[9];
    const float* Wd    = (const float*)d_in[10];
    const float* ln1   = (const float*)d_in[11];
    const float* ln2   = (const float*)d_in[12];
    const float* finln = (const float*)d_in[13];
    const float* fc1w  = (const float*)d_in[14];
    const float* fc1b  = (const float*)d_in[15];
    const float* fc2w  = (const float*)d_in[16];
    const float* fc2b  = (const float*)d_in[17];
    const float* outw  = (const float*)d_in[18];
    const float* outb  = (const float*)d_in[19];

    char* ws = (char*)d_ws;
    float*  x  = (float*)ws;                               // 32 MB
    bf16_t* hb = (bf16_t*)(ws + 33554432);                 // 16 MB
    char* arena = ws + 50331648;
    // attention-phase overlay
    float*  Qf = (float*)(arena);
    float*  Kf = (float*)(arena + 33554432);
    float*  Vf = (float*)(arena + 67108864);
    bf16_t* Qb = (bf16_t*)(arena + 100663296);
    bf16_t* Kb = (bf16_t*)(arena + 117440512);
    bf16_t* Vt = (bf16_t*)(arena + 134217728);
    bf16_t* Ob = (bf16_t*)(arena + 150994944);
    // mlp-phase overlay (same arena)
    float*  G  = (float*)(arena);
    float*  U  = (float*)(arena + 94371840);
    bf16_t* T  = (bf16_t*)(arena + 188743680);

    embed_gather<<<(Mc * Dc) / 256, 256, 0, stream>>>(ids, emb, x);

    dim3 gdd(Dc / 64, Mc / 256);    // 32 x 16
    dim3 gdf(Fc / 64, Mc / 256);    // 88 x 16
    dim3 gattn(8, Hc, Bc);

    for (int i = 0; i < Lc; ++i) {
        const float* wq = Wq + (size_t)i * Dc * Dc;
        const float* wk = Wk + (size_t)i * Dc * Dc;
        const float* wv = Wv + (size_t)i * Dc * Dc;
        const float* wo = Wo + (size_t)i * Dc * Dc;
        const float* wg = Wg + (size_t)i * Dc * Fc;
        const float* wu = Wu + (size_t)i * Dc * Fc;
        const float* wd = Wd + (size_t)i * Fc * Dc;

        rmsnorm_bf16<<<Mc, 256, 0, stream>>>(x, ln1 + (size_t)i * Dc, hb);
        gemm_bf16w32<0><<<gdd, 256, 0, stream>>>(hb, wq, Qf, Mc, Dc, Dc);
        gemm_bf16w32<0><<<gdd, 256, 0, stream>>>(hb, wk, Kf, Mc, Dc, Dc);
        gemm_bf16w32<0><<<gdd, 256, 0, stream>>>(hb, wv, Vf, Mc, Dc, Dc);
        rope_pack<<<(Bc * Sc * Hc * 64) / 256, 256, 0, stream>>>(Qf, Kf, Qb, Kb);
        vpack<<<(Bc * Sc * Hc * HDc) / 256, 256, 0, stream>>>(Vf, Vt);
        attn_flash<<<gattn, 256, 0, stream>>>(Qb, Kb, Vt, amask, Ob);
        gemm_bf16w32<1><<<gdd, 256, 0, stream>>>(Ob, wo, x, Mc, Dc, Dc);

        rmsnorm_bf16<<<Mc, 256, 0, stream>>>(x, ln2 + (size_t)i * Dc, hb);
        gemm_bf16w32<0><<<gdf, 256, 0, stream>>>(hb, wg, G, Mc, Fc, Dc);
        gemm_bf16w32<0><<<gdf, 256, 0, stream>>>(hb, wu, U, Mc, Fc, Dc);
        silu_mul<<<(Mc * Fc) / 256, 256, 0, stream>>>(G, U, T);
        gemm_bf16w32<1><<<gdd, 256, 0, stream>>>(T, wd, x, Mc, Dc, Fc);
    }

    head_kernel<<<1, 256, 0, stream>>>(x, finln, catf, fc1w, fc1b, fc2w, fc2b,
                                       outw, outb, (float*)d_out);
}
